// FCLGCN_Integrated_38551626449135
// MI455X (gfx1250) — matrix operand order repaired and checked by reading, not compiled, not run on hardware
//
#include <hip/hip_runtime.h>
#include <math.h>

typedef __attribute__((ext_vector_type(16))) __bf16 v16bf;
typedef __attribute__((ext_vector_type(8)))  float  v8f;

#define Bsz 512
#define Tsz 32
#define Nsz 64
#define Fsz 32
#define NHn 4
#define Dsz 2048
#define GHsz 64
#define DHsz 512

#define WMMA_BF16(a, b, c) \
  __builtin_amdgcn_wmma_f32_16x16x32_bf16(false, (a), false, (b), (short)0, (c), false, false)

// ---------------- WMMA fragment helpers (wave32, per ISA 7.12.2 layouts) ----------------

// A fragment: 16x32 bf16, row-major source, row stride ld (elements).
// lane<16: M=lane, K = {0..7, 16..23}; lane>=16: M=lane-16, K = {8..15, 24..31}
__device__ __forceinline__ v16bf load_a_frag(const __bf16* src, int ld) {
  int lane = threadIdx.x & 31;
  const __bf16* row = src + (size_t)(lane & 15) * ld;
  int kb = (lane < 16) ? 0 : 8;
  v16bf a;
#pragma unroll
  for (int i = 0; i < 8; ++i) {
    a[i]     = row[kb + i];
    a[8 + i] = row[kb + 16 + i];
  }
  return a;
}

// B fragment: 32x16 bf16; N = lane&15; lane<16: K = {0..7, 16..23}; lane>=16: K = {8..15, 24..31}
// element address: src + K*strideK + N*strideN
__device__ __forceinline__ v16bf load_b_frag(const __bf16* src, int strideK, int strideN) {
  int lane = threadIdx.x & 31;
  const __bf16* p = src + (size_t)(lane & 15) * strideN;
  int kb = (lane < 16) ? 0 : 8;
  v16bf b;
#pragma unroll
  for (int i = 0; i < 8; ++i) {
    b[i]     = p[(size_t)(kb + i) * strideK];
    b[8 + i] = p[(size_t)(kb + 16 + i) * strideK];
  }
  return b;
}

// C/D: VGPR r -> M = mb + r ; N = lane&15 ; mb = (lane<16)?0:8
__device__ __forceinline__ void store_c_bf16(__bf16* dst, int ld, const v8f& c) {
  int lane = threadIdx.x & 31;
  __bf16* p = dst + (lane & 15) + (size_t)((lane < 16) ? 0 : 8) * ld;
#pragma unroll
  for (int r = 0; r < 8; ++r) p[(size_t)r * ld] = (__bf16)c[r];
}

__device__ __forceinline__ void atom_add_c_lds(float* dst, int ld, const v8f& c) {
  int lane = threadIdx.x & 31;
  float* p = dst + (lane & 15) + (size_t)((lane < 16) ? 0 : 8) * ld;
#pragma unroll
  for (int r = 0; r < 8; ++r) atomicAdd(&p[(size_t)r * ld], c[r]);
}

// ---------------- BN1 statistics: per (t,f), stats over (B,N) ----------------
__global__ void bn1_stats_kernel(const float* __restrict__ x, float* mean, float* var) {
  __shared__ float r1[256], r2[256];
  int tf = blockIdx.x;                 // t*32 + f
  int t = tf >> 5, f = tf & 31;
  float s = 0.f, sq = 0.f;
  for (int i = threadIdx.x; i < Bsz * Nsz; i += 256) {
    int b = i >> 6, n = i & 63;
    float v = x[((((size_t)b * Tsz + t) * Nsz + n) << 5) + f];
    s += v; sq += v * v;
  }
  int tid = threadIdx.x;
  r1[tid] = s; r2[tid] = sq; __syncthreads();
  for (int st = 128; st > 0; st >>= 1) {
    if (tid < st) { r1[tid] += r1[tid + st]; r2[tid] += r2[tid + st]; }
    __syncthreads();
  }
  if (tid == 0) {
    float m = r1[0] / 32768.f;
    mean[tf] = m;
    var[tf]  = r2[0] / 32768.f - m * m;
  }
}

// ---------------- Laplacian + Chebyshev basis per t (note 2L/2 - I == -D^-1/2 A D^-1/2) --------
__global__ void cheby_kernel(const float* __restrict__ A, __bf16* __restrict__ cheb) {
  __shared__ float sT[64 * 64];   // raw A, then Lhat
  __shared__ float sA[64 * 64];   // A'', then T2
  __shared__ float dinv[64];
  int t = blockIdx.x, n = threadIdx.x;   // 64 threads
  const float* At = A + (size_t)t * 4096;
  for (int m = 0; m < 64; ++m) sT[n * 64 + m] = At[n * 64 + m];
  __syncthreads();
  float rowsum = 0.f;
  for (int m = 0; m < 64; ++m) {
    float a = (n == m) ? 0.f : (fmaxf(sT[n * 64 + m], 0.f) + fmaxf(sT[m * 64 + n], 0.f));
    sA[n * 64 + m] = a;
    rowsum += a;
  }
  dinv[n] = rsqrtf(rowsum + 1e-10f);
  __syncthreads();
  for (int m = 0; m < 64; ++m) sT[n * 64 + m] = -dinv[n] * sA[n * 64 + m] * dinv[m];
  __syncthreads();
  __bf16* c0 = cheb + (size_t)t * 4 * 4096;
  for (int m = 0; m < 64; ++m) c0[n * 64 + m] = (__bf16)((n == m) ? 1.f : 0.f);     // T0 = I
  for (int m = 0; m < 64; ++m) c0[4096 + n * 64 + m] = (__bf16)sT[n * 64 + m];      // T1 = L
  for (int m = 0; m < 64; ++m) {                                                    // T2 = 2LL - I
    float acc = 0.f;
    for (int j = 0; j < 64; ++j) acc += sT[n * 64 + j] * sT[j * 64 + m];
    float v = 2.f * acc - ((n == m) ? 1.f : 0.f);
    sA[n * 64 + m] = v;
    c0[2 * 4096 + n * 64 + m] = (__bf16)v;
  }
  __syncthreads();
  for (int m = 0; m < 64; ++m) {                                                    // T3 = 2 L T2 - T1
    float acc = 0.f;
    for (int j = 0; j < 64; ++j) acc += sT[n * 64 + j] * sA[j * 64 + m];
    c0[3 * 4096 + n * 64 + m] = (__bf16)(2.f * acc - sT[n * 64 + m]);
  }
}

// ---------------- fused Chebynet: per (t,b): res = relu(sum_k (T_k @ bn1(x)) @ W_k) ----------
__global__ void chebynet_kernel(const float* __restrict__ x, const __bf16* __restrict__ cheb,
                                const float* __restrict__ chebW,
                                const float* __restrict__ m1, const float* __restrict__ v1,
                                const float* __restrict__ g1, const float* __restrict__ b1,
                                __bf16* __restrict__ res, float* bn2_sum, float* bn2_sumsq) {
  __shared__ __bf16 sX[64 * 32];        // bn1-normalized x, bf16
  __shared__ __bf16 sW[4 * 32 * 32];    // cheb_W for this t, bf16
  __shared__ __bf16 sZ[4 * 64 * 32];    // per-wave T_k @ X
  __shared__ float  sR[64 * 32];        // fp32 accumulator
  int t = blockIdx.x, b = blockIdx.y;
  int wave = threadIdx.x >> 5;          // wave == k
  for (int e = threadIdx.x; e < 2048; e += 128) {
    int n = e >> 5, f = e & 31;
    float xv = x[((((size_t)b * Tsz + t) * Nsz + n) << 5) + f];
    int tf = (t << 5) + f;
    float nv = (xv - m1[tf]) * rsqrtf(v1[tf] + 1e-5f) * g1[tf] + b1[tf];
    sX[e] = (__bf16)nv;
    sR[e] = 0.f;
  }
  for (int e = threadIdx.x; e < 4096; e += 128)
    sW[e] = (__bf16)chebW[(size_t)t * 4096 + e];
  __syncthreads();

  // stage A: Z_k = T_k(64x64) @ X(64x32)
  const __bf16* Ck = cheb + ((size_t)t * 4 + wave) * 4096;
  v8f zacc[4][2] = {};
#pragma unroll
  for (int kk = 0; kk < 2; ++kk) {
    v16bf af[4], bx[2];
#pragma unroll
    for (int mi = 0; mi < 4; ++mi) af[mi] = load_a_frag(Ck + (mi * 16) * 64 + kk * 32, 64);
#pragma unroll
    for (int nj = 0; nj < 2; ++nj) bx[nj] = load_b_frag(sX + (kk * 32) * 32 + nj * 16, 32, 1);
#pragma unroll
    for (int mi = 0; mi < 4; ++mi)
#pragma unroll
      for (int nj = 0; nj < 2; ++nj) zacc[mi][nj] = WMMA_BF16(af[mi], bx[nj], zacc[mi][nj]);
  }
  __bf16* Zk = sZ + wave * 2048;
#pragma unroll
  for (int mi = 0; mi < 4; ++mi)
#pragma unroll
    for (int nj = 0; nj < 2; ++nj) store_c_bf16(Zk + (mi * 16) * 32 + nj * 16, 32, zacc[mi][nj]);

  // stage B: R_k = Z_k(64x32) @ W_k(32x32), accumulate into sR
  v16bf bw[2];
#pragma unroll
  for (int nj = 0; nj < 2; ++nj) bw[nj] = load_b_frag(sW + wave * 1024 + nj * 16, 32, 1);
#pragma unroll
  for (int mi = 0; mi < 4; ++mi) {
    v16bf az = load_a_frag(Zk + (mi * 16) * 32, 32);
#pragma unroll
    for (int nj = 0; nj < 2; ++nj) {
      v8f racc = {};
      racc = WMMA_BF16(az, bw[nj], racc);
      atom_add_c_lds(sR + (mi * 16) * 32 + nj * 16, 32, racc);
    }
  }
  __syncthreads();

  // epilogue: relu, store res (bf16), BN2 partial sums
  __bf16* rb = res + (((size_t)t * Bsz + b) << 11);
  for (int e = threadIdx.x; e < 2048; e += 128)
    rb[e] = (__bf16)fmaxf(sR[e], 0.f);
  if (threadIdx.x < 32) {
    int f = threadIdx.x;
    float s = 0.f, sq = 0.f;
    for (int n = 0; n < 64; ++n) {
      float r = fmaxf(sR[n * 32 + f], 0.f);
      s += r; sq += r * r;
    }
    atomicAdd(&bn2_sum[(t << 5) + f], s);
    atomicAdd(&bn2_sumsq[(t << 5) + f], sq);
  }
}

// ---------------- BN2 apply + sigmoid; re-layout [T,B,N,F] -> y_seq [B,T,D] bf16 -----------
__global__ void bn2_apply_kernel(const __bf16* __restrict__ res,
                                 const float* __restrict__ sum, const float* __restrict__ sumsq,
                                 const float* __restrict__ g, const float* __restrict__ bb,
                                 __bf16* __restrict__ y) {
  size_t idx = (size_t)blockIdx.x * 256 + threadIdx.x;     // 2^25 elements
  int f = idx & 31;
  int n = (idx >> 5) & 63;
  int b = (idx >> 11) & 511;
  int t = (int)(idx >> 20);
  int tf = (t << 5) + f;
  float m = sum[tf] / 32768.f;
  float v = sumsq[tf] / 32768.f - m * m;
  float r = (float)res[idx];
  float yv = (r - m) * rsqrtf(v + 1e-5f) * g[tf] + bb[tf];
  yv = 1.f / (1.f + __expf(-yv));
  y[(((size_t)b * Tsz + t) << 11) + (n << 5) + f] = (__bf16)yv;
}

// ---------------- generic bf16 WMMA GEMM: C[M,N] = A[M,K] @ B[K,N] ----------------
template <typename OutT>
__global__ void gemm_bf16_kernel(const __bf16* __restrict__ A, const __bf16* __restrict__ Bm,
                                 OutT* __restrict__ C, int M, int N, int K) {
  __shared__ __bf16 sA[128 * 32];
  __shared__ __bf16 sB[32 * 128];
  int bm = blockIdx.y * 128, bn = blockIdx.x * 128;
  int wave = threadIdx.x >> 5;
  int wm = (wave >> 1) * 32;     // 0,32,64,96
  int wn = (wave & 1) * 64;      // 0,64
  v8f acc[2][4] = {};
  for (int k0 = 0; k0 < K; k0 += 32) {
    {
      int r = threadIdx.x >> 1, c = (threadIdx.x & 1) * 16;
      int gm = bm + r;
      const __bf16* src = A + (size_t)gm * K + k0 + c;
#pragma unroll
      for (int i = 0; i < 16; ++i) sA[r * 32 + c + i] = (gm < M) ? src[i] : (__bf16)0.f;
    }
    {
      int r = threadIdx.x >> 3, c = (threadIdx.x & 7) * 16;
      const __bf16* src = Bm + (size_t)(k0 + r) * N + bn + c;
#pragma unroll
      for (int i = 0; i < 16; ++i) sB[r * 128 + c + i] = (bn + c + i < N) ? src[i] : (__bf16)0.f;
    }
    if (k0 + 32 < K) {   // speculative prefetch of next K-slab -> global_prefetch_b8
      int gm = bm + (threadIdx.x >> 1);
      if (gm < M) __builtin_prefetch(A + (size_t)gm * K + k0 + 32, 0, 1);
      __builtin_prefetch(Bm + (size_t)(k0 + 32 + (threadIdx.x >> 3)) * N + bn, 0, 1);
    }
    __syncthreads();
    v16bf af[2], bfg[4];
    af[0] = load_a_frag(sA + (wm + 0) * 32, 32);
    af[1] = load_a_frag(sA + (wm + 16) * 32, 32);
#pragma unroll
    for (int j = 0; j < 4; ++j) bfg[j] = load_b_frag(sB + wn + j * 16, 128, 1);
#pragma unroll
    for (int i = 0; i < 2; ++i)
#pragma unroll
      for (int j = 0; j < 4; ++j) acc[i][j] = WMMA_BF16(af[i], bfg[j], acc[i][j]);
    __syncthreads();
  }
  int lane = threadIdx.x & 31;
  int cn = lane & 15, cmb = (lane < 16) ? 0 : 8;
#pragma unroll
  for (int i = 0; i < 2; ++i)
#pragma unroll
    for (int j = 0; j < 4; ++j) {
      int gn = bn + wn + j * 16 + cn;
      if (gn >= N) continue;
      int gm0 = bm + wm + i * 16 + cmb;
#pragma unroll
      for (int r = 0; r < 8; ++r) {
        int gm = gm0 + r;
        if (gm < M) C[(size_t)gm * N + gn] = (OutT)acc[i][j][r];
      }
    }
}

// ---------------- attention: per (h,b): softmax(qk^T * dh^-0.5) @ v ----------------
__global__ void attn_kernel(const __bf16* __restrict__ q, const __bf16* __restrict__ k,
                            const __bf16* __restrict__ v, __bf16* __restrict__ ctx) {
  __shared__ float  sS[32 * 32];
  __shared__ __bf16 sP[32 * 32];
  int h = blockIdx.x, b = blockIdx.y;
  int wave = threadIdx.x >> 5;
  const __bf16* qb = q + ((size_t)b * Tsz) * Dsz + h * DHsz;
  const __bf16* kb = k + ((size_t)b * Tsz) * Dsz + h * DHsz;
  const __bf16* vb = v + ((size_t)b * Tsz) * Dsz + h * DHsz;
  __bf16* cb = ctx + ((size_t)b * Tsz) * Dsz + h * DHsz;
  for (int e = threadIdx.x; e < 1024; e += 128) sS[e] = 0.f;
  __syncthreads();

  // scores: M=T=32, N=T=32, K=dh=512 split across 4 waves
  v8f sacc[2][2] = {};
  int kc = wave * 128;
#pragma unroll
  for (int kk = 0; kk < 4; ++kk) {
    int ko = kc + kk * 32;
    v16bf a0 = load_a_frag(qb + ko, Dsz);
    v16bf a1 = load_a_frag(qb + (size_t)16 * Dsz + ko, Dsz);
    v16bf b0 = load_b_frag(kb + ko, 1, Dsz);                      // B[kd][t'] = k[t'][kd]
    v16bf b1 = load_b_frag(kb + (size_t)16 * Dsz + ko, 1, Dsz);
    sacc[0][0] = WMMA_BF16(a0, b0, sacc[0][0]);
    sacc[0][1] = WMMA_BF16(a0, b1, sacc[0][1]);
    sacc[1][0] = WMMA_BF16(a1, b0, sacc[1][0]);
    sacc[1][1] = WMMA_BF16(a1, b1, sacc[1][1]);
  }
  atom_add_c_lds(sS, 32, sacc[0][0]);
  atom_add_c_lds(sS + 16, 32, sacc[0][1]);
  atom_add_c_lds(sS + 16 * 32, 32, sacc[1][0]);
  atom_add_c_lds(sS + 16 * 32 + 16, 32, sacc[1][1]);
  __syncthreads();

  if (threadIdx.x < 32) {
    int r = threadIdx.x;
    float scale = rsqrtf((float)DHsz);
    float mx = -1e30f;
    for (int c = 0; c < 32; ++c) mx = fmaxf(mx, sS[r * 32 + c] * scale);
    float ssum = 0.f;
    for (int c = 0; c < 32; ++c) {
      float ev = __expf(sS[r * 32 + c] * scale - mx);
      sS[r * 32 + c] = ev;
      ssum += ev;
    }
    float inv = 1.f / ssum;
    for (int c = 0; c < 32; ++c) sP[r * 32 + c] = (__bf16)(sS[r * 32 + c] * inv);
  }
  __syncthreads();

  // ctx: M=32, K=32, N=512 split across 4 waves (128 cols each)
  int nb = wave * 128;
  v16bf a0 = load_a_frag(sP, 32);
  v16bf a1 = load_a_frag(sP + 16 * 32, 32);
#pragma unroll
  for (int nj = 0; nj < 8; ++nj) {
    int col = nb + nj * 16;
    v16bf bv = load_b_frag(vb + col, Dsz, 1);                     // B[t'][n] = v[t'][col+n]
    v8f c0 = {}, c1 = {};
    c0 = WMMA_BF16(a0, bv, c0);
    c1 = WMMA_BF16(a1, bv, c1);
    store_c_bf16(cb + col, Dsz, c0);
    store_c_bf16(cb + (size_t)16 * Dsz + col, Dsz, c1);
  }
}

// ---------------- residual + LayerNorm: h = LN(y + ctxWf + bf) ----------------
__global__ void ln_kernel(const __bf16* __restrict__ y, const __bf16* __restrict__ ao,
                          const float* __restrict__ bfb, const float* __restrict__ g,
                          const float* __restrict__ bb, __bf16* __restrict__ hout) {
  __shared__ float red[256];
  int row = blockIdx.x;                        // b*T + t
  int tid = threadIdx.x;
  const __bf16* yr = y + (size_t)row * Dsz;
  const __bf16* ar = ao + (size_t)row * Dsz;
  float vals[8];
  float s = 0.f;
#pragma unroll
  for (int i = 0; i < 8; ++i) {
    int d = tid + i * 256;
    float hv = (float)yr[d] + (float)ar[d] + bfb[d];
    vals[i] = hv;
    s += hv;
  }
  red[tid] = s; __syncthreads();
  for (int st = 128; st > 0; st >>= 1) { if (tid < st) red[tid] += red[tid + st]; __syncthreads(); }
  float mean = red[0] / (float)Dsz; __syncthreads();
  float sq = 0.f;
#pragma unroll
  for (int i = 0; i < 8; ++i) { float dd = vals[i] - mean; sq += dd * dd; }
  red[tid] = sq; __syncthreads();
  for (int st = 128; st > 0; st >>= 1) { if (tid < st) red[tid] += red[tid + st]; __syncthreads(); }
  float inv = rsqrtf(red[0] / (float)Dsz + 1e-5f);
  __bf16* hr = hout + (size_t)row * Dsz;
#pragma unroll
  for (int i = 0; i < 8; ++i) {
    int d = tid + i * 256;
    hr[d] = (__bf16)((vals[i] - mean) * inv * g[d] + bb[d]);
  }
}

// ---------------- GRU over T; 4 batch rows per block ----------------
__global__ void gru_kernel(const float* __restrict__ gi, const float* __restrict__ Whh,
                           const float* __restrict__ bih, const float* __restrict__ bhh,
                           float* __restrict__ ygru) {
  __shared__ float sW[192 * 64];
  __shared__ float sh[4][64];
  __shared__ float sbih[192], sbhh[192];
  int tid = threadIdx.x;
  int bl = tid >> 6, n = tid & 63;
  int bb = blockIdx.x * 4 + bl;
  for (int e = tid; e < 192 * 64; e += 256) sW[e] = Whh[e];
  if (tid < 192) { sbih[tid] = bih[tid]; sbhh[tid] = bhh[tid]; }
  sh[bl][n] = 0.f;
  __syncthreads();
  for (int t = 0; t < Tsz; ++t) {
    const float* gir = gi + ((size_t)bb * Tsz + t) * 192;
    float ir = gir[n] + sbih[n];
    float iz = gir[64 + n] + sbih[64 + n];
    float in = gir[128 + n] + sbih[128 + n];
    float hr = sbhh[n], hz = sbhh[64 + n], hn = sbhh[128 + n];
    const float* hp = sh[bl];
    for (int j = 0; j < 64; ++j) {
      float hj = hp[j];
      hr += sW[n * 64 + j] * hj;
      hz += sW[(64 + n) * 64 + j] * hj;
      hn += sW[(128 + n) * 64 + j] * hj;
    }
    float r  = 1.f / (1.f + __expf(-(ir + hr)));
    float zg = 1.f / (1.f + __expf(-(iz + hz)));
    float ng = tanhf(in + r * hn);
    float hnew = (1.f - zg) * ng + zg * hp[n];
    __syncthreads();
    sh[bl][n] = hnew;
    ygru[((size_t)bb * Tsz + t) * 64 + n] = 1.f / (1.f + __expf(-hnew));
    __syncthreads();
  }
}

// ---------------- heads: proj (normalized) + classifier ----------------
__global__ void head_kernel(const float* __restrict__ of, const float* __restrict__ linW,
                            const float* __restrict__ linb, const float* __restrict__ clsW,
                            const float* __restrict__ clsb, float* __restrict__ out) {
  __shared__ float sx[2048];
  __shared__ float red[128];
  int b = blockIdx.x, tid = threadIdx.x;   // 128 threads
  const float* row = of + (size_t)b * 2048;
  for (int e = tid; e < 2048; e += 128) sx[e] = row[e];
  __syncthreads();
  const float* w = linW + (size_t)tid * 2048;
  float dot = 0.f;
  for (int i = 0; i < 2048; ++i) dot += sx[i] * w[i];
  float p = 1.f / (1.f + __expf(-(dot + linb[tid])));
  red[tid] = p * p; __syncthreads();
  for (int st = 64; st > 0; st >>= 1) { if (tid < st) red[tid] += red[tid + st]; __syncthreads(); }
  float nr = fmaxf(sqrtf(red[0]), 1e-12f); __syncthreads();
  out[(size_t)b * 128 + tid] = p / nr;
  for (int c = 0; c < 3; ++c) {
    const float* cw = clsW + (size_t)c * 2048;
    float part = 0.f;
    for (int i = tid; i < 2048; i += 128) part += sx[i] * cw[i];
    red[tid] = part; __syncthreads();
    for (int st = 64; st > 0; st >>= 1) { if (tid < st) red[tid] += red[tid + st]; __syncthreads(); }
    if (tid == 0) out[65536 + (size_t)b * 3 + c] = 1.f / (1.f + __expf(-(red[0] + clsb[c])));
    __syncthreads();
  }
}

// ---------------- conversions ----------------
__global__ void f2bf_kernel(const float* __restrict__ s, __bf16* __restrict__ d, int n) {
  int i = blockIdx.x * 256 + threadIdx.x;
  if (i < n) d[i] = (__bf16)s[i];
}
__global__ void wih_t_kernel(const float* __restrict__ w, __bf16* __restrict__ wt) {
  int idx = blockIdx.x * 256 + threadIdx.x;    // 2048*192
  if (idx >= 2048 * 192) return;
  int j = idx % 192, d = idx / 192;
  wt[idx] = (__bf16)w[(size_t)j * 2048 + d];
}

// ---------------- host-side launch ----------------
extern "C" void kernel_launch(void* const* d_in, const int* in_sizes, int n_in,
                              void* d_out, int out_size, void* d_ws, size_t ws_size,
                              hipStream_t stream) {
  const float* x      = (const float*)d_in[0];
  const float* Amat   = (const float*)d_in[1];
  const float* chebW  = (const float*)d_in[2];
  const float* bn1_g  = (const float*)d_in[3];
  const float* bn1_b  = (const float*)d_in[4];
  const float* bn2_g  = (const float*)d_in[5];
  const float* bn2_b  = (const float*)d_in[6];
  const float* Wq     = (const float*)d_in[7];
  const float* Wk     = (const float*)d_in[8];
  const float* Wv     = (const float*)d_in[9];
  const float* Wf     = (const float*)d_in[10];
  const float* bfb    = (const float*)d_in[11];
  const float* ln_g   = (const float*)d_in[12];
  const float* ln_b   = (const float*)d_in[13];
  const float* W_ih   = (const float*)d_in[14];
  const float* W_hh   = (const float*)d_in[15];
  const float* b_ih   = (const float*)d_in[16];
  const float* b_hh   = (const float*)d_in[17];
  const float* lin_W  = (const float*)d_in[18];
  const float* lin_b  = (const float*)d_in[19];
  const float* cls_W  = (const float*)d_in[20];
  const float* cls_b  = (const float*)d_in[21];
  float* out = (float*)d_out;

  char* wsp = (char*)d_ws;
  auto alloc = [&](size_t bytes) -> char* {
    char* p = wsp;
    wsp += (bytes + 255) & ~(size_t)255;
    return p;
  };
  float*  bn1_mean = (float*)alloc(1024 * 4);
  float*  bn1_var  = (float*)alloc(1024 * 4);
  float*  bn2_sum  = (float*)alloc(1024 * 4);
  float*  bn2_sq   = (float*)alloc(1024 * 4);
  __bf16* cheb_bf  = (__bf16*)alloc((size_t)Tsz * 4 * 4096 * 2);
  __bf16* res_bf   = (__bf16*)alloc((size_t)33554432 * 2);
  __bf16* y_bf     = (__bf16*)alloc((size_t)33554432 * 2);
  __bf16* q_bf     = (__bf16*)alloc((size_t)33554432 * 2);
  __bf16* k_bf     = (__bf16*)alloc((size_t)33554432 * 2);
  __bf16* v_bf     = (__bf16*)alloc((size_t)33554432 * 2);
  __bf16* ctx_bf   = (__bf16*)alloc((size_t)33554432 * 2);
  __bf16* ao_bf    = (__bf16*)alloc((size_t)33554432 * 2);
  __bf16* h_bf     = (__bf16*)alloc((size_t)33554432 * 2);
  __bf16* wq_bf    = (__bf16*)alloc((size_t)Dsz * Dsz * 2);
  __bf16* wk_bf    = (__bf16*)alloc((size_t)Dsz * Dsz * 2);
  __bf16* wv_bf    = (__bf16*)alloc((size_t)Dsz * Dsz * 2);
  __bf16* wf_bf    = (__bf16*)alloc((size_t)Dsz * Dsz * 2);
  __bf16* wih_t    = (__bf16*)alloc((size_t)2048 * 192 * 2);
  float*  gi       = (float*)alloc((size_t)16384 * 192 * 4);
  float*  ygru     = (float*)alloc((size_t)512 * 32 * 64 * 4);

  hipMemsetAsync(bn2_sum, 0, 1024 * 4, stream);
  hipMemsetAsync(bn2_sq, 0, 1024 * 4, stream);

  bn1_stats_kernel<<<1024, 256, 0, stream>>>(x, bn1_mean, bn1_var);
  cheby_kernel<<<Tsz, 64, 0, stream>>>(Amat, cheb_bf);

  int dd = Dsz * Dsz;
  f2bf_kernel<<<(dd + 255) / 256, 256, 0, stream>>>(Wq, wq_bf, dd);
  f2bf_kernel<<<(dd + 255) / 256, 256, 0, stream>>>(Wk, wk_bf, dd);
  f2bf_kernel<<<(dd + 255) / 256, 256, 0, stream>>>(Wv, wv_bf, dd);
  f2bf_kernel<<<(dd + 255) / 256, 256, 0, stream>>>(Wf, wf_bf, dd);
  wih_t_kernel<<<(2048 * 192 + 255) / 256, 256, 0, stream>>>(W_ih, wih_t);

  chebynet_kernel<<<dim3(Tsz, Bsz), 128, 0, stream>>>(
      x, cheb_bf, chebW, bn1_mean, bn1_var, bn1_g, bn1_b, res_bf, bn2_sum, bn2_sq);
  bn2_apply_kernel<<<131072, 256, 0, stream>>>(res_bf, bn2_sum, bn2_sq, bn2_g, bn2_b, y_bf);

  dim3 gg(Dsz / 128, (Bsz * Tsz) / 128);    // 16 x 128
  gemm_bf16_kernel<__bf16><<<gg, 256, 0, stream>>>(y_bf, wq_bf, q_bf, Bsz * Tsz, Dsz, Dsz);
  gemm_bf16_kernel<__bf16><<<gg, 256, 0, stream>>>(y_bf, wk_bf, k_bf, Bsz * Tsz, Dsz, Dsz);
  gemm_bf16_kernel<__bf16><<<gg, 256, 0, stream>>>(y_bf, wv_bf, v_bf, Bsz * Tsz, Dsz, Dsz);

  attn_kernel<<<dim3(NHn, Bsz), 128, 0, stream>>>(q_bf, k_bf, v_bf, ctx_bf);

  gemm_bf16_kernel<__bf16><<<gg, 256, 0, stream>>>(ctx_bf, wf_bf, ao_bf, Bsz * Tsz, Dsz, Dsz);

  ln_kernel<<<Bsz * Tsz, 256, 0, stream>>>(y_bf, ao_bf, bfb, ln_g, ln_b, h_bf);

  gemm_bf16_kernel<float><<<dim3(2, 128), 256, 0, stream>>>(h_bf, wih_t, gi, Bsz * Tsz, 192, Dsz);

  gru_kernel<<<Bsz / 4, 256, 0, stream>>>(gi, W_hh, b_ih, b_hh, ygru);

  head_kernel<<<Bsz, 128, 0, stream>>>(ygru, lin_W, lin_b, cls_W, cls_b, out);
}
